// AttentionModule_22265110462678
// MI455X (gfx1250) — matrix operand order repaired and checked by reading, not compiled, not run on hardware
//
#include <hip/hip_runtime.h>
#include <hip/hip_bf16.h>

// Shapes from the reference
#define BB 64    // batch
#define DD 64    // input dim (layer 0)
#define NN 1024  // sequence
#define RR 128   // hidden rank
#define HH 64    // final output channels
#define CC 128   // padded channel stride for activation buffers

typedef unsigned short u16;
typedef __attribute__((ext_vector_type(16))) __bf16 bf16x16;
typedef __attribute__((ext_vector_type(16))) u16    u16x16;
typedef __attribute__((ext_vector_type(8)))  u16    u16x8;
typedef __attribute__((ext_vector_type(8)))  float  f32x8;

__device__ __forceinline__ u16 f2bf(float f) {   // native v_cvt bf16 path on gfx1250
  return __builtin_bit_cast(u16, (__bf16)f);
}

// A-fragment: per-lane two contiguous 8-element K runs (k0..k0+7, k0+16..k0+23)
__device__ __forceinline__ bf16x16 mk_frag(u16x8 a, u16x8 b) {
  u16x16 t;
#pragma unroll
  for (int i = 0; i < 8; ++i) { t[i] = a[i]; t[i + 8] = b[i]; }
  return __builtin_bit_cast(bf16x16, t);
}

__device__ __forceinline__ f32x8 wmma_bf16(bf16x16 a, bf16x16 b, f32x8 c) {
  return __builtin_amdgcn_wmma_f32_16x16x32_bf16(false, a, false, b, (short)0, c, false, false);
}

__device__ __forceinline__ f32x8 zero8() {
  f32x8 z;
#pragma unroll
  for (int i = 0; i < 8; ++i) z[i] = 0.f;
  return z;
}

// ---------------------------------------------------------------------------
// Kernel 0: f32 -> bf16 weight conversion (runs once per launch; 320KB total)
// ---------------------------------------------------------------------------
__global__ void cvt_kernel(const float* __restrict__ s, u16* __restrict__ d, int n) {
  int i = blockIdx.x * blockDim.x + threadIdx.x;
  if (i < n) d[i] = f2bf(s[i]);
}

// ---------------------------------------------------------------------------
// Kernel 1: per-batch scale = rsqrt(#neighbors+1), and x (B,D,N) -> inpT (B,N,C) bf16
// ---------------------------------------------------------------------------
__global__ void prep_kernel(const float* __restrict__ x, const int* __restrict__ L,
                            float* __restrict__ sinv, u16* __restrict__ inpT) {
  int b = blockIdx.x, tid = threadIdx.x;
  __shared__ int red[8];
  int cnt = 0;
  for (int n = tid; n < NN; n += 256) cnt += (L[b * NN + n] >= 1) ? 1 : 0;
#pragma unroll
  for (int off = 16; off; off >>= 1) cnt += __shfl_xor(cnt, off, 32);
  if ((tid & 31) == 0) red[tid >> 5] = cnt;
  __syncthreads();
  if (tid == 0) {
    int s = 0;
#pragma unroll
    for (int i = 0; i < 8; ++i) s += red[i];
    sinv[b] = rsqrtf((float)s + 1.0f);
  }
  for (int i = tid; i < DD * NN; i += 256) {
    int d = i / NN, n = i % NN;
    inpT[((size_t)b * NN + n) * CC + d] = f2bf(x[((size_t)b * DD + d) * NN + n]);
  }
}

// ---------------------------------------------------------------------------
// Kernel 2: Q/K/V = sigmoid(W @ inp).  One wave per 16(r)x16(n) tile; one inpT
// B-fragment feeds 3 WMMAs.  Weights already bf16 row-major -> pure 16B loads.
// Qt,Kt stored (N,R); V stored (R,N). All bf16.
// ---------------------------------------------------------------------------
__global__ void proj_kernel(const u16* __restrict__ Wq, const u16* __restrict__ Wk,
                            const u16* __restrict__ Wv, int Din,
                            const u16* __restrict__ inpT,
                            u16* __restrict__ Qt, u16* __restrict__ Kt, u16* __restrict__ Vb) {
  int lane = threadIdx.x & 31, wid = threadIdx.x >> 5;
  int l16 = lane & 15, half = lane >> 4;
  int rt = wid, nt = blockIdx.x, b = blockIdx.y;

  f32x8 cq = zero8(), ck = zero8(), cv = zero8();
  const u16* inprow = inpT + ((size_t)b * NN + nt * 16 + l16) * CC;  // B col n = l16
  int row = rt * 16 + l16;                                           // A row r = l16
  int nkc = Din >> 5;
  for (int kc = 0; kc < nkc; ++kc) {
    bf16x16 Bf = mk_frag(*(const u16x8*)(inprow + kc * 32 + half * 8), *(const u16x8*)(inprow + kc * 32 + half * 8 + 16));
    int wbase = row * Din + kc * 32 + half * 8;
    bf16x16 Aq = mk_frag(*(const u16x8*)(Wq + wbase), *(const u16x8*)(Wq + wbase + 16));
    bf16x16 Ak = mk_frag(*(const u16x8*)(Wk + wbase), *(const u16x8*)(Wk + wbase + 16));
    bf16x16 Av = mk_frag(*(const u16x8*)(Wv + wbase), *(const u16x8*)(Wv + wbase + 16));
    cq = wmma_bf16(Aq, Bf, cq);
    ck = wmma_bf16(Ak, Bf, ck);
    cv = wmma_bf16(Av, Bf, cv);
  }
  // C-frag (v,lane): r = rt*16 + v + half*8, n = nt*16 + l16
  size_t nrow = (size_t)b * NN + nt * 16 + l16;
  u16x8 q8, k8;
#pragma unroll
  for (int v = 0; v < 8; ++v) {
    q8[v] = f2bf(1.f / (1.f + __expf(-cq[v])));
    k8[v] = f2bf(1.f / (1.f + __expf(-ck[v])));
  }
  *(u16x8*)(Qt + nrow * RR + rt * 16 + half * 8) = q8;   // contiguous 16B store
  *(u16x8*)(Kt + nrow * RR + rt * 16 + half * 8) = k8;
#pragma unroll
  for (int v = 0; v < 8; ++v) {
    float sv = 1.f / (1.f + __expf(-cv[v]));
    Vb[((size_t)b * RR + rt * 16 + half * 8 + v) * NN + nt * 16 + l16] = f2bf(sv);
  }
}

// ---------------------------------------------------------------------------
// Kernel 3: fused flash attention + output projection + SiLU (+mask).
// One wave per (batch, 16-query tile).  O accumulator = 8 f32 C-frags (16n x 128r).
// ---------------------------------------------------------------------------
template <bool LAST>
__global__ __launch_bounds__(32)
void attn_kernel(const u16* __restrict__ Qt, const u16* __restrict__ Kt,
                 const u16* __restrict__ Vb, const u16* __restrict__ Wo,
                 const int* __restrict__ L, const float* __restrict__ sinv,
                 u16* __restrict__ outT, float* __restrict__ outF, int Cout) {
  int lane = threadIdx.x & 31;
  int l16 = lane & 15, half = lane >> 4;
  int nt = blockIdx.x, b = blockIdx.y;
  float si = sinv[b];

  __shared__ u16 pbuf[16 * 32];    // P tile (C-layout -> A-layout transpose)
  __shared__ u16 obuf[16 * 128];   // normalized O tile for the Wo GEMM

  // Q A-fragments: constant across the whole key loop (A[n][r] from Qt (N,R))
  const u16* qrow = Qt + ((size_t)b * NN + nt * 16 + l16) * RR;
  bf16x16 AQ[4];
#pragma unroll
  for (int rc = 0; rc < 4; ++rc)
    AQ[rc] = mk_frag(*(const u16x8*)(qrow + rc * 32 + half * 8),
                     *(const u16x8*)(qrow + rc * 32 + half * 8 + 16));

  f32x8 O[8];
#pragma unroll
  for (int f = 0; f < 8; ++f) O[f] = zero8();
  float mrun[8], lrun[8];
#pragma unroll
  for (int v = 0; v < 8; ++v) { mrun[v] = -1e30f; lrun[v] = 0.f; }

  for (int mc = 0; mc < NN / 32; ++mc) {        // 32 keys per chunk
    f32x8 s0 = zero8(), s1 = zero8();
    const u16* kr0 = Kt + ((size_t)b * NN + mc * 32 + l16) * RR;  // B col m = l16
    const u16* kr1 = kr0 + 16 * RR;                               // cols m+16..m+31
#pragma unroll
    for (int rc = 0; rc < 4; ++rc) {
      int kof = rc * 32 + half * 8;
      bf16x16 B0 = mk_frag(*(const u16x8*)(kr0 + kof), *(const u16x8*)(kr0 + kof + 16));
      bf16x16 B1 = mk_frag(*(const u16x8*)(kr1 + kof), *(const u16x8*)(kr1 + kof + 16));
      s0 = wmma_bf16(AQ[rc], B0, s0);
      s1 = wmma_bf16(AQ[rc], B1, s1);
    }
    // online softmax; row n lives at (v, lane-half): reduce across 16-lane halves
    float c[8];
#pragma unroll
    for (int v = 0; v < 8; ++v) {
      float a0 = s0[v] * si, a1 = s1[v] * si;
      float t = fmaxf(a0, a1);
      for (int off = 1; off < 16; off <<= 1) t = fmaxf(t, __shfl_xor(t, off, 32));
      float mn = fmaxf(mrun[v], t);
      c[v] = __expf(mrun[v] - mn);
      mrun[v] = mn;
      float p0 = __expf(a0 - mn), p1 = __expf(a1 - mn);
      float rs = p0 + p1;
      for (int off = 1; off < 16; off <<= 1) rs += __shfl_xor(rs, off, 32);
      lrun[v] = lrun[v] * c[v] + rs;
      s0[v] = p0; s1[v] = p1;
    }
#pragma unroll
    for (int f = 0; f < 8; ++f)
#pragma unroll
      for (int v = 0; v < 8; ++v) O[f][v] *= c[v];
    // transpose P through LDS: C-layout -> A-layout
    __syncthreads();
#pragma unroll
    for (int v = 0; v < 8; ++v) {
      pbuf[(v + half * 8) * 32 + l16]      = f2bf(s0[v]);
      pbuf[(v + half * 8) * 32 + 16 + l16] = f2bf(s1[v]);
    }
    __syncthreads();
    bf16x16 P = mk_frag(*(const u16x8*)(pbuf + l16 * 32 + half * 8),
                        *(const u16x8*)(pbuf + l16 * 32 + half * 8 + 16));
    // O[n][r] += P[n][m] * V[r][m]   (B-frag contiguous from V (R,N))
#pragma unroll
    for (int rf = 0; rf < 8; ++rf) {
      const u16* vr = Vb + ((size_t)b * RR + rf * 16 + l16) * NN + mc * 32 + half * 8;
      bf16x16 BV = mk_frag(*(const u16x8*)vr, *(const u16x8*)(vr + 16));
      O[rf] = wmma_bf16(P, BV, O[rf]);
    }
  }

  // normalize and stage O (16 x 128) in LDS as bf16 for the output projection
  float inv[8];
#pragma unroll
  for (int v = 0; v < 8; ++v) inv[v] = 1.f / lrun[v];
  __syncthreads();
#pragma unroll
  for (int f = 0; f < 8; ++f)
#pragma unroll
    for (int v = 0; v < 8; ++v)
      obuf[(v + half * 8) * 128 + f * 16 + l16] = f2bf(O[f][v] * inv[v]);
  __syncthreads();

  // out[n][c] = silu( sum_r O[n][r] * Wo[c][r] )   (B2 = two 16B bf16 loads)
  int nct = Cout >> 4;
  for (int ct = 0; ct < nct; ++ct) {
    f32x8 acc = zero8();
    const u16* wrow = Wo + (ct * 16 + l16) * RR;
#pragma unroll
    for (int rc = 0; rc < 4; ++rc) {
      bf16x16 A2 = mk_frag(*(const u16x8*)(obuf + l16 * 128 + rc * 32 + half * 8),
                           *(const u16x8*)(obuf + l16 * 128 + rc * 32 + half * 8 + 16));
      bf16x16 B2 = mk_frag(*(const u16x8*)(wrow + rc * 32 + half * 8), *(const u16x8*)(wrow + rc * 32 + half * 8 + 16));
      acc = wmma_bf16(A2, B2, acc);
    }
    if (LAST) {
#pragma unroll
      for (int v = 0; v < 8; ++v) {            // n = v+half*8 -> contiguous f32 stores
        int n = nt * 16 + v + half * 8;
        float y = acc[v];
        y = y / (1.f + __expf(-y));
        outF[((size_t)b * HH + ct * 16 + l16) * NN + n] = y;
      }
    } else {
#pragma unroll
      for (int v = 0; v < 8; ++v) {
        int n = nt * 16 + v + half * 8;
        float y = acc[v];
        y = y / (1.f + __expf(-y));
        y *= (float)L[b * NN + n];             // mask = Lrow (0/1)
        outT[((size_t)b * NN + n) * CC + ct * 16 + l16] = f2bf(y);
      }
    }
  }
}

// ---------------------------------------------------------------------------
extern "C" void kernel_launch(void* const* d_in, const int* in_sizes, int n_in,
                              void* d_out, int out_size, void* d_ws, size_t ws_size,
                              hipStream_t stream) {
  (void)in_sizes; (void)n_in; (void)out_size; (void)ws_size;
  const float* x      = (const float*)d_in[0];
  const int*   L      = (const int*)d_in[1];
  const float* wq0    = (const float*)d_in[2];
  const float* wqr    = (const float*)d_in[3];
  const float* wk0    = (const float*)d_in[4];
  const float* wkr    = (const float*)d_in[5];
  const float* wv0    = (const float*)d_in[6];
  const float* wvr    = (const float*)d_in[7];
  const float* wor    = (const float*)d_in[8];
  const float* wolast = (const float*)d_in[9];
  float* out = (float*)d_out;

  // workspace: 1KB scales + 5 x 16.78MB bf16 activation buffers + 320KB bf16 weights
  char* w = (char*)d_ws;
  float* sinv = (float*)w;
  u16* inpT0 = (u16*)(w + 1024);
  size_t cbuf = (size_t)BB * NN * CC;
  u16* inpT1 = inpT0 + cbuf;
  u16* Qt = inpT1 + cbuf;
  u16* Kt = Qt + cbuf;
  u16* Vb = Kt + cbuf;
  u16* wq0b = Vb + cbuf;              // R*D
  u16* wk0b = wq0b + RR * DD;
  u16* wv0b = wk0b + RR * DD;
  u16* wqrb = wv0b + RR * DD;         // 2*R*R each
  u16* wkrb = wqrb + 2 * RR * RR;
  u16* wvrb = wkrb + 2 * RR * RR;
  u16* worb = wvrb + 2 * RR * RR;
  u16* wolb = worb + 2 * RR * RR;     // H*R

  // one-time bf16 weight conversion
  cvt_kernel<<<(RR * DD + 255) / 256, 256, 0, stream>>>(wq0, wq0b, RR * DD);
  cvt_kernel<<<(RR * DD + 255) / 256, 256, 0, stream>>>(wk0, wk0b, RR * DD);
  cvt_kernel<<<(RR * DD + 255) / 256, 256, 0, stream>>>(wv0, wv0b, RR * DD);
  cvt_kernel<<<(2 * RR * RR + 255) / 256, 256, 0, stream>>>(wqr, wqrb, 2 * RR * RR);
  cvt_kernel<<<(2 * RR * RR + 255) / 256, 256, 0, stream>>>(wkr, wkrb, 2 * RR * RR);
  cvt_kernel<<<(2 * RR * RR + 255) / 256, 256, 0, stream>>>(wvr, wvrb, 2 * RR * RR);
  cvt_kernel<<<(2 * RR * RR + 255) / 256, 256, 0, stream>>>(wor, worb, 2 * RR * RR);
  cvt_kernel<<<(HH * RR + 255) / 256, 256, 0, stream>>>(wolast, wolb, HH * RR);

  dim3 gP(NN / 16, BB), bP(256);   // 8 waves/block = the 8 r-tiles
  dim3 gA(NN / 16, BB), bA(32);    // 1 wave per query tile; 4096 waves

  prep_kernel<<<dim3(BB), dim3(256), 0, stream>>>(x, L, sinv, inpT0);

  // layer 0
  proj_kernel<<<gP, bP, 0, stream>>>(wq0b, wk0b, wv0b, DD, inpT0, Qt, Kt, Vb);
  attn_kernel<false><<<gA, bA, 0, stream>>>(Qt, Kt, Vb, worb, L, sinv, inpT1, nullptr, CC);
  // layer 1
  proj_kernel<<<gP, bP, 0, stream>>>(wqrb, wkrb, wvrb, CC, inpT1, Qt, Kt, Vb);
  attn_kernel<false><<<gA, bA, 0, stream>>>(Qt, Kt, Vb, worb + RR * RR, L, sinv, inpT0, nullptr, CC);
  // layer 2 (last): output projection writes f32 straight to d_out
  proj_kernel<<<gP, bP, 0, stream>>>(wqrb + RR * RR, wkrb + RR * RR, wvrb + RR * RR, CC, inpT0, Qt, Kt, Vb);
  attn_kernel<true><<<gA, bA, 0, stream>>>(Qt, Kt, Vb, wolb, L, sinv, nullptr, out, HH);
}
